// W4A16Linear_66829691126216
// MI455X (gfx1250) — matrix operand order repaired and checked by reading, not compiled, not run on hardware
//
#include <hip/hip_runtime.h>

typedef _Float16 half_t;
typedef __attribute__((ext_vector_type(2)))  _Float16 h2;
typedef __attribute__((ext_vector_type(16))) _Float16 v16h;
typedef __attribute__((ext_vector_type(8)))  _Float16 v8h;
typedef __attribute__((ext_vector_type(8)))  float    v8f;

#define M_DIM 8192
#define K_DIM 4096
#define N_DIM 11008
#define G_SZ  128
#define K_HALF (K_DIM / 2)

#define BM 128
#define BN 128
#define BK 32
#define NIT (K_HALF / BK)    // 64 fused double-k iterations (k and k+K/2 together)
#define NG2 (K_HALF / G_SZ)  // 16 group-pairs (g, g+16)
#define KPG (G_SZ / BK)      // 4 iterations per group-pair

#define XPAD 8
#define XROW (BK + XPAD)     // 40 halves = 80 B x-row: 16B aligned, conflict-free
#define CPAD 8
#define CROW (BN + CPAD)     // 136 halves = 272 B c-row

#define NIB_MASK 0x000F000Fu
#define FP16_MAGIC 0x64006400u   // packed fp16(1024.0); 0x6400|q == fp16(1024+q) exactly

// Epilogue staging aliases the x double-buffer: strictly ordered by barriers.
union Smem {
  half_t xs[2][2][BM * XROW];  // [buf][k-half][row*XROW + k]  (40960 B)
  half_t cs[BM * CROW];        // (34816 B)
};

__device__ __forceinline__ void async_copy_16B(unsigned lds_off, const void* gptr) {
  // VDST = LDS byte address, VADDR = 64-bit global address (GV mode), tracked by ASYNCcnt
  asm volatile("global_load_async_to_lds_b128 %0, %1, off"
               :: "v"(lds_off), "v"((unsigned long long)(size_t)gptr)
               : "memory");
}

union Frag { v16h v; h2 p[8]; };

__global__ __launch_bounds__(256)
void w4a16_wmma_kernel(const half_t* __restrict__ x,
                       const int*    __restrict__ wq,      // [K/2, N] int32 nibble-packed
                       const half_t* __restrict__ scales,  // [K/G, N]
                       const half_t* __restrict__ zeros,   // [K/G, N]
                       half_t*       __restrict__ out) {   // [M, N]
  __shared__ Smem smem;

  const int tid  = threadIdx.x;
  const int lane = tid & 31;
  const int l16  = lane & 15;
  const int hi   = lane >> 4;
  const int wave = tid >> 5;
  const int wn   = wave * 16;         // each wave owns one 16-wide n-tile, full 128-row m

  const int bm = blockIdx.y * BM;
  const int bn = blockIdx.x * BN;

  // Stage = x rows for k-tile it*32 (sub 0) and it*32 + K/2 (sub 1): 1024 x 16B chunks.
  auto issue_stage = [&](int it) {
    const int buf = it & 1;
    const int kb  = it * BK;
#pragma unroll
    for (int i = 0; i < 4; ++i) {
      const int c   = tid + i * 256;
      const int sub = c >> 9;
      const int cc  = c & 511;
      const int row = cc >> 2;
      const int seg = cc & 3;
      const unsigned lds_off = (unsigned)(size_t)(&smem.xs[buf][sub][0]) +
                               (unsigned)(row * (XROW * 2) + seg * 16);
      async_copy_16B(lds_off,
                     x + (size_t)(bm + row) * K_DIM + kb + sub * K_HALF + seg * 8);
    }
  };

  v8f acc[8] = {};   // [m-tile]; D = 16(n) x 16(m) each, one n-tile per wave

  issue_stage(0);
  issue_stage(1);

  const h2 c1024 = { (half_t)1024.0f, (half_t)1024.0f };
  const int ncol_g = bn + wn + l16;   // this lane's weight column

  for (int g = 0; g < NG2; ++g) {
    // Dequant constants for this lane's n column, both k-halves (groups g and g+16).
    h2 s2[2], b2[2];
#pragma unroll
    for (int kh = 0; kh < 2; ++kh) {
      const int gi = kh * NG2 + g;
      const half_t s = scales[(size_t)gi * N_DIM + ncol_g];
      const half_t z = zeros [(size_t)gi * N_DIM + ncol_g];
      const half_t b = (half_t)(-(float)z * (float)s);
      s2[kh] = (h2){ s, s };
      b2[kh] = (h2){ b, b };
    }

    for (int ki = 0; ki < KPG; ++ki) {
      const int it   = g * KPG + ki;
      const int krow = it * BK;   // packed row base; feeds BOTH k and k+K/2

      if (it < NIT - 1) asm volatile("s_wait_asynccnt 0x4" ::: "memory");
      else              asm volatile("s_wait_asynccnt 0x0" ::: "memory");
      __syncthreads();

      const int buf = it & 1;

      // Fragment tile t (0..15): k-half = t>>3, m-tile = t&7.
      // Per lane: 32 contiguous bytes = 2 x ds_load_b128.
      const half_t* fb = &smem.xs[buf][0][0] + l16 * XROW + hi * 16;
      auto ldfrag = [&](int t) -> v16h {
        const half_t* p   = fb + (t >> 3) * (BM * XROW) + (t & 7) * (16 * XROW);
        const v8h     lo8 = *(const v8h*)(p);
        const v8h     hi8 = *(const v8h*)(p + 8);
        return __builtin_shufflevector(lo8, hi8,
                                       0,1,2,3,4,5,6,7,8,9,10,11,12,13,14,15);
      };

      // ---- preload first two B tiles; their latency hides under dequant ----
      v16h cur = ldfrag(0);
      v16h nxt = ldfrag(1);

      // ---- dequantize both nibble halves of each dword into two A fragments ----
      Frag aw[2];  // [k-half]
      {
        const int* col = wq + (size_t)krow * N_DIM + ncol_g;
#pragma unroll
        for (int j = 0; j < 8; ++j) {
          // 16-bit A layout (contiguous, matches B): VGPR j -> k = hi*16 + 2j(+1)
          const int kA = hi * 16 + j * 2;
          const unsigned d0 = (unsigned)col[(size_t)(kA    ) * N_DIM];
          const unsigned d1 = (unsigned)col[(size_t)(kA + 1) * N_DIM];
          // v_perm_b32: gather byte0 of each dword -> t = 0x00BB00AA (0x0C = zero byte)
          const unsigned tt  = __builtin_amdgcn_perm(d1, d0, 0x0C040C00u);
          const unsigned plo = ( tt       & NIB_MASK) | FP16_MAGIC;  // v_and_or_b32
          const unsigned phi = ((tt >> 4) & NIB_MASK) | FP16_MAGIC;  // lshr + v_and_or_b32
          const h2 qlo = __builtin_bit_cast(h2, plo) - c1024;        // exact (v_pk_add)
          const h2 qhi = __builtin_bit_cast(h2, phi) - c1024;        // exact
          aw[0].p[j] = qlo * s2[0] + b2[0];                          // v_pk_fma_f16
          aw[1].p[j] = qhi * s2[1] + b2[1];
        }
      }

      // ---- software-pipelined WMMA stream: tile t consumes loads issued at t-2 ----
#pragma unroll
      for (int t = 0; t < 16; ++t) {
        const v16h use = cur;
        cur = nxt;
        if (t < 14) nxt = ldfrag(t + 2);
        acc[t & 7] = __builtin_amdgcn_wmma_f32_16x16x32_f16(
            false, aw[t >> 3].v, false, use, (short)0, acc[t & 7], false, false);
      }

      __syncthreads();
      if (it + 2 < NIT) issue_stage(it + 2);  // refill buffer everyone just drained
    }
  }

  // ---- epilogue: D is n-major; transpose through LDS for coalesced fp16 stores ----
#pragma unroll
  for (int mt = 0; mt < 8; ++mt) {
    const int mrow = mt * 16 + l16;
#pragma unroll
    for (int rp = 0; rp < 4; ++rp) {
      // D VGPR r: row M' = r + 8*hi (our n), column = lane%16 (our m)
      const int ncol = wn + hi * 8 + 2 * rp;
      union { unsigned u; half_t h[2]; } cv;
      cv.h[0] = (half_t)acc[mt][2 * rp];
      cv.h[1] = (half_t)acc[mt][2 * rp + 1];
      *(unsigned*)&smem.cs[mrow * CROW + ncol] = cv.u;
    }
  }
  __syncthreads();

  // Coalesced flush: 128 rows x 256B = 2048 x 16B chunks, 8 per thread.
#pragma unroll
  for (int i = 0; i < 8; ++i) {
    const int c   = tid + i * 256;
    const int row = c >> 4;
    const int seg = c & 15;
    const uint4 v = *(const uint4*)&smem.cs[row * CROW + seg * 8];
    *(uint4*)&out[(size_t)(bm + row) * N_DIM + bn + seg * 8] = v;
  }
}

extern "C" void kernel_launch(void* const* d_in, const int* in_sizes, int n_in,
                              void* d_out, int out_size, void* d_ws, size_t ws_size,
                              hipStream_t stream) {
  const half_t* x      = (const half_t*)d_in[0];
  const int*    wq     = (const int*)   d_in[1];
  const half_t* scales = (const half_t*)d_in[2];
  const half_t* zeros  = (const half_t*)d_in[3];
  // d_in[4] = group_size scalar (compile-time G_SZ here)
  half_t* out = (half_t*)d_out;

  dim3 grid(N_DIM / BN, M_DIM / BM);   // 86 x 64 blocks
  hipLaunchKernelGGL(w4a16_wmma_kernel, grid, dim3(256), 0, stream,
                     x, wq, scales, zeros, out);
}
